// HyperbolicLayer_33131377722111
// MI455X (gfx1250) — matrix operand order repaired and checked by reading, not compiled, not run on hardware
//
#include <hip/hip_runtime.h>
#include <hip/hip_bf16.h>

// MI455X / gfx1250, wave32. Fused Poincare-ball hyperbolic classifier:
//   stage 1: expmap0(z) -> f16 A matrix (N x 512), z_norm per point
//   stage 2: per-class constants + f16 B matrices (-p, normals)
//   stage 3: TDM-staged (tensor_load_to_lds) double-buffered B tiles in LDS,
//            dual WMMA GEMM (v_wmma_f32_16x16x32_f16) + asinh epilogue fused.

typedef __attribute__((ext_vector_type(16))) _Float16 v16h;
typedef __attribute__((ext_vector_type(8)))  _Float16 v8h;
typedef __attribute__((ext_vector_type(8)))  float    v8f;
typedef __attribute__((ext_vector_type(4)))  unsigned int u32x4;
typedef __attribute__((ext_vector_type(4)))  int          i32x4;
typedef __attribute__((ext_vector_type(8)))  int          i32x8;

#define DCH   512          // channel dim (K)
#define PCLS  256          // classes (N of GEMM)
#define HW    4096         // 64*64
#define NBAT  8
#define NPTS  (NBAT * HW)  // 32768 spatial points (M of GEMM)

// LDS staging: one B tile = 16 class rows x 1024B (f16), padded by TDM to a
// 1040B row stride (pad 16B per 1024B) to rotate LDS banks by 4 words/row.
#define BROW  1040
#define BMAT  (16 * BROW)      // 16640 B per matrix tile
#define BBUF  (2 * BMAT)       // ph tile + nw tile per buffer = 33280 B

#if defined(__has_builtin)
#if __has_builtin(__builtin_amdgcn_tensor_load_to_lds) && \
    __has_builtin(__builtin_amdgcn_s_wait_tensorcnt)
#define USE_TDM 1
#endif
#endif

// ---------------------------------------------------------------------------
// Stage 1: expmap0 over channels. One block per (b,y) row, one lane per x.
// ---------------------------------------------------------------------------
__global__ __launch_bounds__(64) void prep_z_kernel(
    const float* __restrict__ z, _Float16* __restrict__ zs,
    float* __restrict__ znorm) {
  const int by = blockIdx.x;            // b*64 + y
  const int x  = threadIdx.x;           // 0..63
  const float* zp = z + (size_t)(by >> 6) * (DCH * HW) + (by & 63) * 64 + x;

  float acc = 0.f;
  for (int d = 0; d < DCH; ++d) {       // lanes: consecutive x -> coalesced
    float v = zp[(size_t)d * HW];
    acc += v * v;
  }
  const float r  = sqrtf(acc);
  const float rn = fmaxf(r, 1e-15f);
  const float scale = tanhf(rn) / rn;   // expmap0 scale (c = 1)
  const int   m = by * 64 + x;          // flattened b*4096 + y*64 + x
  znorm[m] = scale * scale * acc;       // ||expmap0(u)||^2

  _Float16* orow = zs + (size_t)m * DCH;
  for (int d0 = 0; d0 < DCH; d0 += 8) {
    v8h pk;
#pragma unroll
    for (int j = 0; j < 8; ++j)
      pk[j] = (_Float16)(zp[(size_t)(d0 + j) * HW] * scale);
    *(v8h*)(orow + d0) = pk;            // 16B store per lane
  }
}

// ---------------------------------------------------------------------------
// Stage 2: per-class constants {||p||^2, ||w||^2, -p.w} + f16 B packing.
// ---------------------------------------------------------------------------
__global__ __launch_bounds__(64) void prep_p_kernel(
    const float* __restrict__ p, const float* __restrict__ nrm,
    _Float16* __restrict__ ph, _Float16* __restrict__ nw,
    float* __restrict__ pcn) {
  const int cls = blockIdx.x;
  const int t   = threadIdx.x;
  const float* pr = p   + (size_t)cls * DCH;
  const float* nr = nrm + (size_t)cls * DCH;

  float pn = 0.f, wn = 0.f, pdw = 0.f;
  for (int d = t; d < DCH; d += 64) {
    float pv = pr[d], nv = nr[d];
    pn += pv * pv; wn += nv * nv; pdw += -pv * nv;
  }
  __shared__ float r0[64], r1[64], r2[64];
  r0[t] = pn; r1[t] = wn; r2[t] = pdw;
  __syncthreads();
  for (int s = 32; s > 0; s >>= 1) {
    if (t < s) { r0[t] += r0[t + s]; r1[t] += r1[t + s]; r2[t] += r2[t + s]; }
    __syncthreads();
  }
  if (t == 0) {
    pcn[cls] = r0[0]; pcn[PCLS + cls] = r1[0]; pcn[2 * PCLS + cls] = r2[0];
  }
  const int d0 = t * 8;
  v8h a, b;
#pragma unroll
  for (int j = 0; j < 8; ++j) {
    a[j] = (_Float16)(-pr[d0 + j]);
    b[j] = (_Float16)( nr[d0 + j]);
  }
  *(v8h*)(ph + (size_t)cls * DCH + d0) = a;
  *(v8h*)(nw + (size_t)cls * DCH + d0) = b;
}

// ---------------------------------------------------------------------------
// Helpers for TDM staging
// ---------------------------------------------------------------------------
__device__ __forceinline__ unsigned lds_off(const void* p) {
  // generic LDS pointer = {shared aperture, byte offset}; low 32 bits = offset
  return (unsigned)(unsigned long long)p;
}

#if defined(USE_TDM)
// Issue one TDM 2-D tile load: 16 rows x 1024B contiguous global -> LDS with
// +16B padding per 1024B row (ISA D#: data_size=8B, dims in 8B units).
// This toolchain's builtin takes 6 args:
//   (uint32x4 g0, int32x8 g1, int32x4 g2, int32x4 g3, int32x8 extra, int cpol)
__device__ __forceinline__ void tdm_load_tile(unsigned lds_byte,
                                              const void* gsrc) {
  unsigned long long ga = (unsigned long long)gsrc;
  u32x4 g0;
  g0[0] = 1u;                              // count=1, user-mode descriptor
  g0[1] = lds_byte;                        // lds_addr
  g0[2] = (unsigned)ga;                    // global_addr[31:0]
  g0[3] = (unsigned)(ga >> 32) | (2u << 30);  // global_addr[56:32], type=2
  i32x8 g1;
  g1[0] = (int)((3u << 16)      // data_size = 8B
              | (1u << 20)      // pad_enable
              | (7u << 22)      // pad_interval: every 128 x 8B = 1024B
              | (3u << 25));    // pad_amount: 4 DWORDs = 16B
  g1[1] = (int)(128u << 16);    // tensor_dim0 = 128 (one row = 1024B)
  g1[2] = (int)(16u << 16);     // tensor_dim1 = 16 rows
  g1[3] = (int)(128u << 16);    // tile_dim0 = 128
  g1[4] = (int)(16u | (1u << 16)); // tile_dim1 = 16, tile_dim2 = 1
  g1[5] = 128;                  // tensor_dim0_stride = 128 (contiguous rows)
  g1[6] = 0;
  g1[7] = 0;
  i32x4 g2; g2[0] = 1; g2[1] = 1; g2[2] = 0; g2[3] = 0;  // dim2=1, dim3=1
  i32x4 g3; g3[0] = 0; g3[1] = (int)(1u << 16); g3[2] = 0; g3[3] = 0; // dim4=1
  i32x8 g4 = {0, 0, 0, 0, 0, 0, 0, 0};                   // unused group
  __builtin_amdgcn_tensor_load_to_lds(g0, g1, g2, g3, g4, 0);
}
#endif

// Fallback cooperative copy into the same padded LDS layout (all 256 threads)
__device__ __forceinline__ void stage_tile_fallback(char* dstbuf,
                                                    const _Float16* srcph,
                                                    const _Float16* srcnw,
                                                    int tid) {
#pragma unroll
  for (int i = 0; i < 4; ++i) {
    const int off = i * 4096 + tid * 16;        // 0..16368, 16B chunks
    const int row = off >> 10, col = off & 1023;
    *(v8h*)(dstbuf + row * BROW + col) =
        *(const v8h*)((const char*)srcph + off);
    *(v8h*)(dstbuf + BMAT + row * BROW + col) =
        *(const v8h*)((const char*)srcnw + off);
  }
}

// ---------------------------------------------------------------------------
// Stage 3: fused dual GEMM + hyperbolic-distance epilogue.
// 256 threads = 8 waves; wave -> one 16-point M tile (A cached in 128 VGPRs),
// B tiles TDM-staged & double-buffered in LDS, shared by all 8 waves.
// ---------------------------------------------------------------------------
__global__ __launch_bounds__(256) void hyp_wmma_kernel(
    const _Float16* __restrict__ zs, const float* __restrict__ znorm,
    const _Float16* __restrict__ ph, const _Float16* __restrict__ nw,
    const float* __restrict__ pcn, float* __restrict__ out) {
  __shared__ __align__(16) char bsm[2 * BBUF];  // 66,560 B

  const int tid  = threadIdx.x;
  const int wave = tid >> 5;
  const int lane = tid & 31;
  const int lr   = lane & 15;           // A row / B col / C col within tile
  const int half = lane >> 4;           // K-half selector & C row-half
  const int mtile = blockIdx.x * 8 + wave;
  const int m0    = mtile << 4;

#if defined(USE_TDM)
  if (wave == 0) {                      // prologue: stage ptile 0 into buf 0
    tdm_load_tile(lds_off(bsm), ph);
    tdm_load_tile(lds_off(bsm + BMAT), nw);
  }
#endif

  // --- A fragments for full K=512 (contiguous k = 16*half + s per lane) ---
  const _Float16* ap = zs + (size_t)(m0 + lr) * DCH + half * 16;
  v16h afr[16];
#pragma unroll
  for (int kk = 0; kk < 16; ++kk) {
    union { v16h v; v8h h[2]; } u;
    u.h[0] = *(const v8h*)(ap + kk * 32);
    u.h[1] = *(const v8h*)(ap + kk * 32 + 8);
    afr[kk] = u.v;
  }

  float zn[8];
#pragma unroll
  for (int v = 0; v < 8; ++v) zn[v] = znorm[m0 + 8 * half + v];

  const int bb = m0 >> 12;              // batch index
  const int s0 = m0 & 4095;             // y*64 + x0 within image

#pragma unroll 1
  for (int pt = 0; pt < 16; ++pt) {
#if defined(USE_TDM)
    if (wave == 0) __builtin_amdgcn_s_wait_tensorcnt(0);
    __syncthreads();                    // tile pt visible; prev readers done
    if (wave == 0 && pt < 15) {         // stage tile pt+1 into other buffer
      char* nb = bsm + (((pt + 1) & 1) * BBUF);
      tdm_load_tile(lds_off(nb), ph + (size_t)(pt + 1) * 16 * DCH);
      tdm_load_tile(lds_off(nb + BMAT), nw + (size_t)(pt + 1) * 16 * DCH);
    }
#else
    __syncthreads();                    // prev readers done
    stage_tile_fallback(bsm + (pt & 1) * BBUF, ph + (size_t)pt * 16 * DCH,
                        nw + (size_t)pt * 16 * DCH, tid);
    __syncthreads();                    // tile pt visible
#endif

    const char* cb = bsm + (pt & 1) * BBUF;
    // B fragments (same contiguous k order as A): lane holds column N=lr,
    // K = half*16 .. half*16+15 contiguous -> two 16B ds_load_b128 per step.
    const char* bp = cb + lr * BROW + half * 32;
    const char* bw = bp + BMAT;

    v8f accp = {};                      // p_hat . z
    v8f accw = {};                      // normals . z
#pragma unroll
    for (int kk = 0; kk < 16; ++kk) {
      union { v16h v; v8h h[2]; } up, uw;
      up.h[0] = *(const v8h*)(bp + kk * 64);
      up.h[1] = *(const v8h*)(bp + kk * 64 + 16);
      uw.h[0] = *(const v8h*)(bw + kk * 64);
      uw.h[1] = *(const v8h*)(bw + kk * 64 + 16);
      accp = __builtin_amdgcn_wmma_f32_16x16x32_f16(
          false, afr[kk], false, up.v, (short)0, accp, false, false);
      accw = __builtin_amdgcn_wmma_f32_16x16x32_f16(
          false, afr[kk], false, uw.v, (short)0, accw, false, false);
    }

    const int   prow = (pt << 4) + lr;
    const float pn  = pcn[prow];
    const float wn2 = pcn[PCLS + prow];
    const float pdw = pcn[2 * PCLS + prow];

    float o[8];
#pragma unroll
    for (int v = 0; v < 8; ++v) {
      const float pdz = accp[v];
      const float zdw = accw[v];
      const float znv = zn[v];
      const float denom = fmaxf(1.f + 2.f * pdz + znv * pn, 1e-5f);
      const float alpha = (1.f + 2.f * pdz + znv) / denom;
      const float beta  = (1.f - pn) / denom;
      const float pmyw  = alpha * pdw + beta * zdw;
      const float pmywn = alpha * alpha * pn + 2.f * alpha * beta * pdz +
                          beta * beta * znv;
      const float arg =
          fminf(2.f * pmyw / fmaxf((1.f - pmywn) * wn2, 1e-5f), 85.f);
      o[v] = -(2.f * wn2 * asinhf(arg));
    }
    // lane owns out[b][prow][s0 + half*8 .. +7] -> two coalesced float4s
    float* op = out + ((size_t)bb * PCLS + prow) * HW + s0 + half * 8;
    *(float4*)(op)     = make_float4(o[0], o[1], o[2], o[3]);
    *(float4*)(op + 4) = make_float4(o[4], o[5], o[6], o[7]);
  }
}

// ---------------------------------------------------------------------------
extern "C" void kernel_launch(void* const* d_in, const int* in_sizes, int n_in,
                              void* d_out, int out_size, void* d_ws,
                              size_t ws_size, hipStream_t stream) {
  const float* z   = (const float*)d_in[0];   // (8,512,64,64)
  const float* p   = (const float*)d_in[1];   // (256,512)
  const float* nrm = (const float*)d_in[2];   // (256,512)
  float* out = (float*)d_out;                 // (8,256,64,64)

  // workspace layout (~34.2 MB total)
  char* ws = (char*)d_ws;
  size_t off = 0;
  _Float16* zs = (_Float16*)(ws + off); off += (size_t)NPTS * DCH * 2; // 32 MB
  float* znorm = (float*)(ws + off);    off += (size_t)NPTS * 4;
  _Float16* ph = (_Float16*)(ws + off); off += (size_t)PCLS * DCH * 2;
  _Float16* nw = (_Float16*)(ws + off); off += (size_t)PCLS * DCH * 2;
  float* pcn   = (float*)(ws + off);    off += (size_t)3 * PCLS * 4;
  (void)ws_size; (void)in_sizes; (void)n_in; (void)out_size;

  prep_z_kernel<<<NBAT * 64, 64, 0, stream>>>(z, zs, znorm);
  prep_p_kernel<<<PCLS, 64, 0, stream>>>(p, nrm, ph, nw, pcn);
  hyp_wmma_kernel<<<NPTS / (16 * 8), 256, 0, stream>>>(zs, znorm, ph, nw, pcn,
                                                       out);
}
